// GRU_1073741824033
// MI455X (gfx1250) — compile-verified, matrix operand order checked by reading, not run on hardware
//
#include <hip/hip_runtime.h>
#include <hip/hip_bf16.h>
#include <stdint.h>

typedef __attribute__((ext_vector_type(16))) _Float16 v16h;
typedef __attribute__((ext_vector_type(4)))  _Float16 v4h;
typedef __attribute__((ext_vector_type(8)))  float    v8f;

#define LBV 0.0f
#define UBV 0.2f

// ---------------------------------------------------------------------------
// Convert fp32 weight matrix W (Nrows x K, row stride ldw) into fp16 WMMA
// B-fragment order. Fragment (tileN, tileK) = 512 halves; element (n,k) lives
// at lane*16+elem with lane = ((k&8)?16:0)+n ; elem = (k&7)|((k>>4)<<3).
// Each lane's 16 halves are contiguous (32B) -> coalesced b128 loads later.
// One-off prep kernel; not perf critical.
// ---------------------------------------------------------------------------
__global__ __launch_bounds__(256) void wfrag_prep(const float* __restrict__ W,
                                                  _Float16* __restrict__ dst,
                                                  int N, int K, int ldw, int KT,
                                                  int totalElems) {
  int idx = blockIdx.x * 256 + threadIdx.x;
  if (idx >= totalElems) return;
  int frag   = idx >> 9;
  int within = idx & 511;
  int r  = within >> 5;     // N dim within tile
  int k  = within & 31;     // K dim within tile
  int tileN = frag / KT;
  int tileK = frag - tileN * KT;
  int n  = tileN * 16 + r;
  int kk = tileK * 32 + k;
  float v = (n < N && kk < K) ? W[(size_t)n * ldw + kk] : 0.0f;
  int lane = (((k >> 3) & 1) << 4) + r;
  int elem = (k & 7) | ((k >> 4) << 3);
  dst[(size_t)frag * 512 + lane * 16 + elem] = (_Float16)v;
}

// ---------------------------------------------------------------------------
// C(M,N) = A(M,K) @ W(N,K)^T + bias.  A fp32 row-major (lda, rows 16B aligned),
// W pre-converted to fragment order.  REQUIRES: M % 16 == 0, Ntiles % 8 == 0.
// One block per 16-row M tile covers ALL N tiles (A streamed from HBM once).
// 8 waves; each wave owns Ntiles/8 accumulators.  All B-fragment loads for a
// k-step are issued BEFORE the WMMA chain so waits can drain incrementally.
// No divergent guards around WMMA: EXEC stays all-ones.
// ---------------------------------------------------------------------------
__global__ __launch_bounds__(256) void gemm_wmma(const float* __restrict__ A,
                                                 const _Float16* __restrict__ Bfrag,
                                                 const float* __restrict__ bias,
                                                 float* __restrict__ C,
                                                 int M, int N, int K, int lda, int ldc) {
  __shared__ __align__(32) _Float16 ldsA[512];
  int tid  = threadIdx.x;
  int lane = tid & 31;
  int wave = tid >> 5;
  int mBase  = blockIdx.x * 16;
  int Ntiles = N >> 4;
  int tpw    = Ntiles >> 3;          // tiles per wave (1..3)
  int KT     = (K + 31) >> 5;
  v8f acc[3] = {};
  // staging coords (threads 0..127 stage one float4 each)
  int sr  = tid >> 3;                // 0..15 : row in tile
  int sq  = tid & 7;                 // 0..7  : float4 group
  int skl = sq << 2;                 // k-local 0,4,...,28
  int sln = (((skl >> 3) & 1) << 4) + sr;
  int se0 = (skl & 7) | ((skl >> 4) << 3);
  const float* Arow = A + (size_t)(mBase + sr) * lda + skl;
  const _Float16* Bw = Bfrag + (size_t)wave * KT * 512 + lane * 16;

  for (int kt = 0; kt < KT; ++kt) {
    int k0 = kt << 5;
    if (tid < 128) {
      v4h hv;
      if (k0 + 32 <= K) {            // scalar (SGPR) condition: fast path
        float4 va = *(const float4*)(Arow + k0);
        hv[0] = (_Float16)va.x; hv[1] = (_Float16)va.y;
        hv[2] = (_Float16)va.z; hv[3] = (_Float16)va.w;
      } else {                        // K tail (at most once per K loop)
#pragma unroll
        for (int j = 0; j < 4; ++j) {
          int kk = k0 + skl + j;
          hv[j] = (_Float16)((kk < K) ? Arow[k0 + j] : 0.0f);
        }
      }
      *reinterpret_cast<v4h*>(&ldsA[sln * 16 + se0]) = hv;
    }
    __syncthreads();
    // issue ALL B loads for this k-step first, then the WMMA chain
    v16h b[3];
#pragma unroll
    for (int j = 0; j < 3; ++j) {
      if (j >= tpw) break;
      b[j] = *reinterpret_cast<const v16h*>(&Bw[((size_t)(j << 3) * KT + kt) * 512]);
    }
    v16h a = *reinterpret_cast<const v16h*>(&ldsA[lane * 16]);
#pragma unroll
    for (int j = 0; j < 3; ++j) {
      if (j >= tpw) break;
      acc[j] = __builtin_amdgcn_wmma_f32_16x16x32_f16(false, a, false, b[j],
                                                      (short)0, acc[j], false, false);
    }
    __syncthreads();
  }
  int rBase = mBase + ((lane >> 4) << 3);
#pragma unroll
  for (int j = 0; j < 3; ++j) {
    if (j >= tpw) break;
    int col = ((wave + (j << 3)) << 4) + (lane & 15);
    float bv = (bias != nullptr) ? bias[col] : 0.0f;
#pragma unroll
    for (int r = 0; r < 8; ++r)
      C[(size_t)(rBase + r) * ldc + col] = acc[j][r] + bv;
  }
}

// ---------------------------------------------------------------------------
// GRU recurrence over S steps. Each block privately owns 16 batch rows:
// gh = h @ Whh^T + bhh via WMMA (24 N-tiles, K=128), then fused gate math
// updating h in LDS and restaging it as fp16 A-fragments.  A-fragments are
// hoisted into registers once per step; B loads issued ahead of each chain.
// ---------------------------------------------------------------------------
__global__ __launch_bounds__(256) void gru_recur(const float* __restrict__ gi,
                                                 const _Float16* __restrict__ WhhF,
                                                 const float* __restrict__ bhh,
                                                 float* __restrict__ out, int S) {
  __shared__ float hS[16 * 128];                  // fp32 hidden state
  __shared__ __align__(32) _Float16 aF[4 * 512];  // h as WMMA A-fragments
  __shared__ float ghS[16 * 384];                 // h @ Whh^T + bhh
  int tid  = threadIdx.x;
  int lane = tid & 31;
  int wave = tid >> 5;
  int bBase = blockIdx.x * 16;
  const _Float16* Bw = WhhF + (size_t)(wave * 3) * 4 * 512 + lane * 16;

  for (int i = tid; i < 16 * 128; i += 256) hS[i] = 0.0f;
  for (int i = tid; i < 4 * 512; i += 256) aF[i] = (_Float16)0.0f;
  __syncthreads();

  for (int t = 0; t < S; ++t) {
    // --- gh = h @ Whh^T + bhh (EXEC all-ones; no guards) ----------------
    v16h a[4];
#pragma unroll
    for (int kt = 0; kt < 4; ++kt)
      a[kt] = *reinterpret_cast<const v16h*>(&aF[kt * 512 + lane * 16]);
#pragma unroll
    for (int j = 0; j < 3; ++j) {
      v16h b[4];
#pragma unroll
      for (int kt = 0; kt < 4; ++kt)
        b[kt] = *reinterpret_cast<const v16h*>(&Bw[((size_t)j * 4 + kt) * 512]);
      v8f acc = {};
#pragma unroll
      for (int kt = 0; kt < 4; ++kt)
        acc = __builtin_amdgcn_wmma_f32_16x16x32_f16(false, a[kt], false, b[kt],
                                                     (short)0, acc, false, false);
      int col   = ((wave * 3 + j) << 4) + (lane & 15);
      int rBase = (lane >> 4) << 3;
      float bv  = bhh[col];
#pragma unroll
      for (int r = 0; r < 8; ++r) ghS[(rBase + r) * 384 + col] = acc[r] + bv;
    }
    __syncthreads();
    // --- gates + state update (vectorized) ------------------------------
#pragma unroll
    for (int p = 0; p < 2; ++p) {
      int g   = tid + (p << 8);                   // 0..511
      int row = g >> 5;                           // 0..15
      int q   = g & 31;
      int h0  = q << 2;                           // 0,4,...,124
      int b   = bBase + row;
      const float* gib = gi + ((size_t)b * S + t) * 384;
      float4 gr = *(const float4*)(gib + h0);
      float4 gz = *(const float4*)(gib + 128 + h0);
      float4 gn = *(const float4*)(gib + 256 + h0);
      float4 hr = *(const float4*)(&ghS[row * 384 + h0]);
      float4 hz = *(const float4*)(&ghS[row * 384 + 128 + h0]);
      float4 hn = *(const float4*)(&ghS[row * 384 + 256 + h0]);
      float4 ho = *(const float4*)(&hS[row * 128 + h0]);
      float4 hNew;
#define GATE(c)                                                            \
      {                                                                    \
        float rg = 1.0f / (1.0f + expf(-(gr.c + hr.c)));                   \
        float zg = 1.0f / (1.0f + expf(-(gz.c + hz.c)));                   \
        float ng = tanhf(gn.c + rg * hn.c);                                \
        hNew.c = (1.0f - zg) * ng + zg * ho.c;                             \
      }
      GATE(x) GATE(y) GATE(z) GATE(w)
#undef GATE
      *(float4*)(&hS[row * 128 + h0]) = hNew;
      *(float4*)(out + ((size_t)b * S + t) * 128 + h0) = hNew;
      // restage as A-fragment (4 consecutive k share a lane -> b64 store)
      int kt = h0 >> 5;
      int kl = h0 & 31;
      int ln = (((kl >> 3) & 1) << 4) + row;
      int e0 = (kl & 7) | ((kl >> 4) << 3);
      v4h hv;
      hv[0] = (_Float16)hNew.x; hv[1] = (_Float16)hNew.y;
      hv[2] = (_Float16)hNew.z; hv[3] = (_Float16)hNew.w;
      *reinterpret_cast<v4h*>(&aF[kt * 512 + ln * 16 + e0]) = hv;
    }
    __syncthreads();
  }
}

// ---------------------------------------------------------------------------
// block reductions (blockDim == 256)
// ---------------------------------------------------------------------------
__device__ __forceinline__ float blockReduceSum(float v, float* red) {
  int tid = threadIdx.x;
  red[tid] = v;
  __syncthreads();
  for (int off = 128; off > 0; off >>= 1) {
    if (tid < off) red[tid] += red[tid + off];
    __syncthreads();
  }
  float r = red[0];
  __syncthreads();
  return r;
}
__device__ __forceinline__ float blockReduceMax(float v, float* red) {
  int tid = threadIdx.x;
  red[tid] = v;
  __syncthreads();
  for (int off = 128; off > 0; off >>= 1) {
    if (tid < off) red[tid] = fmaxf(red[tid], red[tid + off]);
    __syncthreads();
  }
  float r = red[0];
  __syncthreads();
  return r;
}

// ---------------------------------------------------------------------------
// Per-batch-row: attention scores + softmax + context, FC, softmax(500),
// 32-iteration rebalance. One block per b.
// gE : (B,S,128) = x @ Wa_x^T + ba      u : (B,128) = h_t @ Wa_h^T
// ---------------------------------------------------------------------------
__global__ __launch_bounds__(256) void finalize_kernel(const float* __restrict__ gE,
                                                       const float* __restrict__ u,
                                                       const float* __restrict__ out1,
                                                       const float* __restrict__ vvec,
                                                       const float* __restrict__ Wfc,
                                                       const float* __restrict__ bfc,
                                                       float* __restrict__ outp, int S) {
  __shared__ float uL[128], vL[128], eS[64], attnS[64], concatL[256];
  __shared__ float oldL[512], wcL[512], red[256];
  int tid  = threadIdx.x;
  int lane = tid & 31;
  int wave = tid >> 5;
  int b = blockIdx.x;

  if (tid < 128) {
    uL[tid] = u[(size_t)b * 128 + tid];
    vL[tid] = vvec[tid];
  }
  __syncthreads();

  // e[s] = sum_h v[h] * tanh(u[h] + gE[b,s,h])  (one wave per s)
  for (int s = wave; s < S; s += 8) {
    float p = 0.0f;
    const float* ge = gE + ((size_t)b * S + s) * 128;
    for (int h = lane; h < 128; h += 32) p += vL[h] * tanhf(uL[h] + ge[h]);
    for (int off = 16; off > 0; off >>= 1) p += __shfl_down(p, off, 32);
    if (lane == 0) eS[s] = p;
  }
  __syncthreads();

  // softmax over S (tiny, serial in thread 0)
  if (tid == 0) {
    float m = eS[0];
    for (int s = 1; s < S; ++s) m = fmaxf(m, eS[s]);
    float sum = 0.0f;
    for (int s = 0; s < S; ++s) { float ev = expf(eS[s] - m); attnS[s] = ev; sum += ev; }
    float inv = 1.0f / sum;
    for (int s = 0; s < S; ++s) attnS[s] *= inv;
  }
  __syncthreads();

  // context + concat(h_t, context)
  if (tid < 128) {
    float c = 0.0f;
    for (int s = 0; s < S; ++s) c += attnS[s] * out1[((size_t)b * S + s) * 128 + tid];
    concatL[tid]       = out1[((size_t)b * S + (S - 1)) * 128 + tid];  // h_t
    concatL[128 + tid] = c;
  }
  __syncthreads();

  // logits[n] = bfc[n] + Wfc[n,:] . concat  (n < 500)
  for (int n = tid; n < 500; n += 256) {
    float acc = bfc[n];
    const float* wr = Wfc + (size_t)n * 256;
#pragma unroll 4
    for (int j = 0; j < 256; ++j) acc += wr[j] * concatL[j];
    oldL[n] = acc;
  }
  __syncthreads();

  // softmax over 500
  float pm = -3.4e38f;
  for (int n = tid; n < 500; n += 256) pm = fmaxf(pm, oldL[n]);
  pm = blockReduceMax(pm, red);
  float ps = 0.0f;
  for (int n = tid; n < 500; n += 256) { float e = expf(oldL[n] - pm); wcL[n] = e; ps += e; }
  ps = blockReduceSum(ps, red);
  float inv = 1.0f / ps;
  for (int n = tid; n < 500; n += 256) {
    float w = wcL[n] * inv;
    oldL[n] = w;                                 // old = w
    wcL[n]  = fminf(fmaxf(w, LBV), UBV);         // wc = clip(w)
  }
  __syncthreads();

  // rebalance, up to 32 iterations (uniform early exit == frozen state)
  for (int it = 0; it < 32; ++it) {
    float pl = 0.0f, pn = 0.0f;
    for (int n = tid; n < 500; n += 256) {
      float wcv = wcL[n];
      pl += oldL[n] - wcv;
      if (wcv != UBV) pn += wcv;
    }
    float leftover = blockReduceSum(pl, red);
    float nsum     = blockReduceSum(pn, red);
    float ns = (nsum == 0.0f) ? 1.0f : nsum;
    float pover = 0.0f;
    for (int n = tid; n < 500; n += 256) {
      float wcv = wcL[n];
      float wc2 = (wcv != UBV) ? wcv + leftover * wcv / ns : wcv;
      oldL[n] = wc2;                             // old = wc2
      if (wc2 > UBV) pover = 1.0f;
    }
    float anyOver = blockReduceSum(pover, red);
    int new_done = (anyOver == 0.0f);
    for (int n = tid; n < 500; n += 256) {
      float wc2 = oldL[n];
      wcL[n] = new_done ? wc2 : fminf(fmaxf(wc2, LBV), UBV);
    }
    __syncthreads();
    if (new_done) break;                         // uniform across block
  }

  for (int n = tid; n < 500; n += 256) outp[(size_t)b * 500 + n] = wcL[n];
}

// ---------------------------------------------------------------------------
extern "C" void kernel_launch(void* const* d_in, const int* in_sizes, int n_in,
                              void* d_out, int out_size, void* d_ws, size_t ws_size,
                              hipStream_t stream) {
  (void)in_sizes; (void)n_in; (void)out_size; (void)ws_size;
  const float* x    = (const float*)d_in[0];
  const float* Wih0 = (const float*)d_in[1];
  const float* Whh0 = (const float*)d_in[2];
  const float* bih0 = (const float*)d_in[3];
  const float* bhh0 = (const float*)d_in[4];
  const float* Wih1 = (const float*)d_in[5];
  const float* Whh1 = (const float*)d_in[6];
  const float* bih1 = (const float*)d_in[7];
  const float* bhh1 = (const float*)d_in[8];
  const float* Wa   = (const float*)d_in[9];
  const float* ba   = (const float*)d_in[10];
  const float* vv   = (const float*)d_in[11];
  const float* Wfc  = (const float*)d_in[12];
  const float* bfc  = (const float*)d_in[13];
  float* outp = (float*)d_out;

  constexpr int B = 1024, S = 60, N = 500, H = 128, G3 = 384;

  uint8_t* base = (uint8_t*)d_ws;
  size_t off = 0;
  auto carve = [&](size_t bytes) {
    void* p = base + off;
    off = (off + bytes + 255) & ~(size_t)255;
    return p;
  };
  float* gi   = (float*)carve((size_t)B * S * G3 * 4);   // shared by layer0/layer1
  float* out0 = (float*)carve((size_t)B * S * H * 4);
  float* out1 = (float*)carve((size_t)B * S * H * 4);
  float* gEb  = (float*)carve((size_t)B * S * H * 4);
  float* ub   = (float*)carve((size_t)B * H * 4);
  _Float16* Wih0F = (_Float16*)carve((size_t)24 * 16 * 512 * 2);
  _Float16* Whh0F = (_Float16*)carve((size_t)24 * 4 * 512 * 2);
  _Float16* Wih1F = (_Float16*)carve((size_t)24 * 4 * 512 * 2);
  _Float16* Whh1F = (_Float16*)carve((size_t)24 * 4 * 512 * 2);
  _Float16* WaxF  = (_Float16*)carve((size_t)8 * 16 * 512 * 2);
  _Float16* WahF  = (_Float16*)carve((size_t)8 * 4 * 512 * 2);

  auto prep = [&](const float* W, _Float16* dst, int n, int k, int ldw) {
    int KT = (k + 31) >> 5;
    int NT = (n + 15) >> 4;
    int total = NT * KT * 512;
    wfrag_prep<<<(total + 255) / 256, 256, 0, stream>>>(W, dst, n, k, ldw, KT, total);
  };
  prep(Wih0,   Wih0F, G3, N, N);
  prep(Whh0,   Whh0F, G3, H, H);
  prep(Wih1,   Wih1F, G3, H, H);
  prep(Whh1,   Whh1F, G3, H, H);
  prep(Wa + H, WaxF,  H,  N, H + N);   // Wa_x = Wa[:, H:H+N]
  prep(Wa,     WahF,  H,  H, H + N);   // Wa_h = Wa[:, 0:H]

  auto gemm = [&](const float* A, const _Float16* Bf, const float* bias, float* C,
                  int M, int Nn, int K, int lda, int ldc) {
    gemm_wmma<<<dim3(M / 16), 256, 0, stream>>>(A, Bf, bias, C, M, Nn, K, lda, ldc);
  };

  // layer 0
  gemm(x, Wih0F, bih0, gi, B * S, G3, N, N, G3);
  gru_recur<<<B / 16, 256, 0, stream>>>(gi, Whh0F, bhh0, out0, S);
  // layer 1 (gi buffer reused)
  gemm(out0, Wih1F, bih1, gi, B * S, G3, H, H, G3);
  gru_recur<<<B / 16, 256, 0, stream>>>(gi, Whh1F, bhh1, out1, S);
  // attention projections
  gemm(x, WaxF, ba, gEb, B * S, H, N, N, H);
  gemm(out1 + (size_t)(S - 1) * H, WahF, nullptr, ub, B, H, H, S * H, H);
  // attention + FC + softmax + rebalance
  finalize_kernel<<<B, 256, 0, stream>>>(gEb, ub, out1, vv, Wfc, bfc, outp, S);
}
